// ThreatGNNv2_36060545417860
// MI455X (gfx1250) — matrix operand order repaired and checked by reading, not compiled, not run on hardware
//
#include <hip/hip_runtime.h>
#include <hip/hip_bf16.h>

typedef __attribute__((ext_vector_type(16))) _Float16 v16h;
typedef __attribute__((ext_vector_type(8)))  float    v8f;

#define BN_RSQRT 0.99999499f  /* rsqrt(1+1e-5) folded at compile time (approx), recomputed in-kernel anyway */

// ---------------------------------------------------------------- utilities
__global__ void k_zero(float* __restrict__ p, int n) {
    int i = blockIdx.x * blockDim.x + threadIdx.x;
    if (i < n) p[i] = 0.0f;
}

// ---------------------------------------------------------------- projection: h = relu(bn(x@Wp + b))
__global__ void k_proj(const float* __restrict__ x, const float* __restrict__ W,
                       const float* __restrict__ b, const float* __restrict__ g,
                       const float* __restrict__ bb, float* __restrict__ h, int n) {
    int i = blockIdx.x * blockDim.x + threadIdx.x;
    if (i >= n * 128) return;
    int node = i >> 7, o = i & 127;
    const float* xr = x + (size_t)node * 16;
    float s = b[o];
#pragma unroll
    for (int k = 0; k < 16; ++k) s = fmaf(xr[k], W[k * 128 + o], s);
    s = s * (g[o] * rsqrtf(1.0f + 1e-5f)) + bb[o];
    h[i] = fmaxf(s, 0.0f);
}

// ---------------------------------------------------------------- pack W[K,nout] fp32 -> WMMA B-tile f16 layout
// packed[((nt*4+kc)*32 + lane)*16 + half] = W[kc*32 + (lane<16?0:8) + (half<8?half:half+8)][nt*16 + lane%16]
__global__ void k_pack_w(const float* __restrict__ W, _Float16* __restrict__ Wp,
                         int ntiles, int nout) {
    int idx = blockIdx.x * blockDim.x + threadIdx.x;
    if (idx >= ntiles * 2048) return;
    int hh   = idx & 15;
    int lane = (idx >> 4) & 31;
    int kc   = (idx >> 9) & 3;
    int nt   = idx >> 11;
    int K = kc * 32 + ((lane < 16) ? 0 : 8) + ((hh < 8) ? hh : hh + 8);
    int n = nt * 16 + (lane & 15);
    Wp[idx] = (_Float16)W[K * nout + n];
}

// ---------------------------------------------------------------- WMMA GEMM: Y[nrows, NT*16] = X[nrows,128] @ Wp + bias
// optional: folded BN (bng/bnb), relu, per-row scale (GCN norm). One wave per 16-row tile.
template <int NT>
__global__ void k_gemm128(const float* __restrict__ X, const _Float16* __restrict__ Wp,
                          const float* __restrict__ bias,
                          const float* __restrict__ bng, const float* __restrict__ bnb,
                          const float* __restrict__ rowscale,
                          float* __restrict__ Y, int nrows, int do_relu) {
    const int lane = threadIdx.x & 31;
    const int wave = threadIdx.x >> 5;
    const int tile = blockIdx.x * 4 + wave;
    if (tile * 16 >= nrows) return;               // wave-uniform guard, EXEC stays all-ones
    const int row = tile * 16 + (lane & 15);
    const int kb  = (lane < 16) ? 0 : 8;

    v8f acc[NT];
#pragma unroll
    for (int nt = 0; nt < NT; ++nt)
        acc[nt] = v8f{0.f, 0.f, 0.f, 0.f, 0.f, 0.f, 0.f, 0.f};

#pragma unroll
    for (int kc = 0; kc < 4; ++kc) {
        const float4* xa = (const float4*)(X + (size_t)row * 128 + kc * 32 + kb);
        const float4* xb = (const float4*)(X + (size_t)row * 128 + kc * 32 + kb + 16);
        float4 a0 = xa[0], a1 = xa[1], c0 = xb[0], c1 = xb[1];
        v16h av;
        av[0]  = (_Float16)a0.x; av[1]  = (_Float16)a0.y; av[2]  = (_Float16)a0.z; av[3]  = (_Float16)a0.w;
        av[4]  = (_Float16)a1.x; av[5]  = (_Float16)a1.y; av[6]  = (_Float16)a1.z; av[7]  = (_Float16)a1.w;
        av[8]  = (_Float16)c0.x; av[9]  = (_Float16)c0.y; av[10] = (_Float16)c0.z; av[11] = (_Float16)c0.w;
        av[12] = (_Float16)c1.x; av[13] = (_Float16)c1.y; av[14] = (_Float16)c1.z; av[15] = (_Float16)c1.w;
#pragma unroll
        for (int nt = 0; nt < NT; ++nt) {
            v16h bv = *(const v16h*)(Wp + ((size_t)(nt * 4 + kc) * 32 + lane) * 16);
            acc[nt] = __builtin_amdgcn_wmma_f32_16x16x32_f16(
                false, av, false, bv, (short)0, acc[nt], false, false);
        }
    }

    const float bnr  = rsqrtf(1.0f + 1e-5f);
    const int   madd = (lane >> 4) << 3;
#pragma unroll
    for (int nt = 0; nt < NT; ++nt) {
        int   col = nt * 16 + (lane & 15);
        float cb  = bias[col];
        float gsc = bng ? bng[col] * bnr : 0.0f;
        float gbb = bnb ? bnb[col] : 0.0f;
#pragma unroll
        for (int r = 0; r < 8; ++r) {
            int   m = tile * 16 + madd + r;
            float v = acc[nt][r] + cb;
            if (bng) v = v * gsc + gbb;
            if (do_relu) v = fmaxf(v, 0.0f);
            if (rowscale) v *= rowscale[m];
            Y[(size_t)m * (NT * 16) + col] = v;
        }
    }
}

// ---------------------------------------------------------------- GCN helpers
__global__ void k_deg(const int* __restrict__ dst, float* __restrict__ deg, int E) {
    int e = blockIdx.x * blockDim.x + threadIdx.x;
    if (e < E) atomicAdd(&deg[dst[e]], 1.0f);
}

__global__ void k_norm(float* __restrict__ deg, int n) {
    int i = blockIdx.x * blockDim.x + threadIdx.x;
    if (i < n) deg[i] = rsqrtf(fmaxf(deg[i], 1.0f));
}

// agg[dst] += lin[src] * norm[src]   (lin already carries norm[src] once — faithful double-apply)
__global__ void k_gcn_scatter(const int* __restrict__ src, const int* __restrict__ dst,
                              const float* __restrict__ lin, const float* __restrict__ norm,
                              float* __restrict__ agg, int E) {
    int gid = blockIdx.x * blockDim.x + threadIdx.x;
    int e = gid >> 5, t = gid & 31;
    if (e >= E) return;
    int   s  = src[e], d = dst[e];
    float ns = norm[s];
    const float* lr = lin + (size_t)s * 128;
    float* ar = agg + (size_t)d * 128;
#pragma unroll
    for (int j = 0; j < 4; ++j) {
        int dim = t + j * 32;
        atomicAdd(&ar[dim], lr[dim] * ns);
    }
}

// h = bn(relu(agg)) + h   (residual)
__global__ void k_post(const float* __restrict__ agg, const float* __restrict__ g,
                       const float* __restrict__ bb, float* __restrict__ h, int n) {
    int i = blockIdx.x * blockDim.x + threadIdx.x;
    if (i >= n * 128) return;
    int   d = i & 127;
    float v = fmaxf(agg[i], 0.0f);
    v = v * (g[d] * rsqrtf(1.0f + 1e-5f)) + bb[d];
    h[i] = v + h[i];
}

// ---------------------------------------------------------------- GAT helpers
__global__ void k_gat_scores(const float* __restrict__ lin, const float* __restrict__ as,
                             const float* __restrict__ ad, float* __restrict__ ssrc,
                             float* __restrict__ sdst, int n) {
    int i = blockIdx.x * blockDim.x + threadIdx.x;
    if (i >= n * 4) return;
    int node = i >> 2, hh = i & 3;
    const float* hr = lin + (size_t)node * 128 + hh * 32;
    float s0 = 0.f, s1 = 0.f;
#pragma unroll
    for (int d = 0; d < 32; ++d) {
        float v = hr[d];
        s0 = fmaf(v, as[d], s0);
        s1 = fmaf(v, ad[d], s1);
    }
    ssrc[i] = s0;
    sdst[i] = s1;
}

__global__ void k_gat_edge1(const int* __restrict__ src, const int* __restrict__ dst,
                            const float* __restrict__ ssrc, const float* __restrict__ sdst,
                            float* __restrict__ denom, int E) {
    int gid = blockIdx.x * blockDim.x + threadIdx.x;
    int e = gid >> 2, hh = gid & 3;
    if (e >= E) return;
    int   s = src[e], d = dst[e];
    float a = ssrc[s * 4 + hh] + sdst[d * 4 + hh];
    a = (a > 0.0f) ? a : 0.2f * a;              // leaky_relu(.,0.2)
    atomicAdd(&denom[d * 4 + hh], expf(a));
}

__global__ void k_gat_edge2(const int* __restrict__ src, const int* __restrict__ dst,
                            const float* __restrict__ lin, const float* __restrict__ ssrc,
                            const float* __restrict__ sdst, const float* __restrict__ denom,
                            float* __restrict__ agg, int E) {
    int gid = blockIdx.x * blockDim.x + threadIdx.x;
    int e = gid >> 5, t = gid & 31;
    if (e >= E) return;
    int s = src[e], d = dst[e];
    const float* lr = lin + (size_t)s * 128;
    float* ar = agg + (size_t)d * 128;
#pragma unroll
    for (int hh = 0; hh < 4; ++hh) {
        float a = ssrc[s * 4 + hh] + sdst[d * 4 + hh];
        a = (a > 0.0f) ? a : 0.2f * a;
        float alpha = a / (denom[d * 4 + hh] + 1e-8f);   // faithful: attn (not exp) / sum exp
        atomicAdd(&ar[hh * 32 + t], lr[hh * 32 + t] * alpha);
    }
}

// ---------------------------------------------------------------- classifier tail
__global__ void k_cls3(const float* __restrict__ c2, const float* __restrict__ W,
                       const float* __restrict__ b, float* __restrict__ out, int n) {
    int node = blockIdx.x * blockDim.x + threadIdx.x;
    if (node >= n) return;
    const float* cr = c2 + (size_t)node * 64;
    float a0 = b[0], a1 = b[1], a2 = b[2];
#pragma unroll 8
    for (int k = 0; k < 64; ++k) {
        float v = cr[k];
        a0 = fmaf(v, W[k * 3 + 0], a0);
        a1 = fmaf(v, W[k * 3 + 1], a1);
        a2 = fmaf(v, W[k * 3 + 2], a2);
    }
    out[(size_t)node * 3 + 0] = a0;
    out[(size_t)node * 3 + 1] = a1;
    out[(size_t)node * 3 + 2] = a2;
}

// ---------------------------------------------------------------- graph readout
__global__ void k_mean(const float* __restrict__ h, float* __restrict__ gsum, int n) {
    int d  = threadIdx.x;          // 128 threads
    int r0 = blockIdx.x * 64;
    float s = 0.0f;
    for (int r = 0; r < 64; ++r) {
        int node = r0 + r;
        if (node < n) s += h[(size_t)node * 128 + d];
    }
    atomicAdd(&gsum[d], s);
}

__global__ void k_gshead(const float* __restrict__ gsum, const float* __restrict__ W1,
                         const float* __restrict__ b1, const float* __restrict__ W2,
                         const float* __restrict__ b2, float* __restrict__ out, int n) {
    __shared__ float s1[64];
    int j = threadIdx.x;           // 64 threads
    float acc = b1[j];
    float inv = 1.0f / (float)n;
    for (int d = 0; d < 128; ++d) acc = fmaf(gsum[d] * inv, W1[d * 64 + j], acc);
    s1[j] = fmaxf(acc, 0.0f);
    __syncthreads();
    if (j == 0) {
        float a = b2[0];
        for (int k = 0; k < 64; ++k) a = fmaf(s1[k], W2[k], a);
        out[0] = 1.0f / (1.0f + expf(-a));
    }
}

// ================================================================ host
extern "C" void kernel_launch(void* const* d_in, const int* in_sizes, int n_in,
                              void* d_out, int out_size, void* d_ws, size_t ws_size,
                              hipStream_t stream) {
    const float* x  = (const float*)d_in[0];
    const int*   ei = (const int*)d_in[1];
    const int N = in_sizes[0] / 16;
    const int E = in_sizes[1] / 2;
    const int* src  = ei;
    const int* dstp = ei + E;

    auto F = [&](int i) { return (const float*)d_in[i]; };
    // insertion-order flattening of params dict
    const float* pW = F(2); const float* pB = F(3); const float* pG = F(4); const float* pBb = F(5);
    const float *lW[4], *lB[4], *lG[4], *lBb[4], *lAs[4], *lAd[4];
    int idx = 6;
    for (int i = 0; i < 4; ++i) {
        lW[i] = F(idx++); lB[i] = F(idx++); lG[i] = F(idx++); lBb[i] = F(idx++);
        lAs[i] = nullptr; lAd[i] = nullptr;
        if (i & 1) { lAs[i] = F(idx++); lAd[i] = F(idx++); }
    }
    const float* c1W = F(idx++); const float* c1B = F(idx++);
    const float* cbG = F(idx++); const float* cbB = F(idx++);
    const float* c2W = F(idx++); const float* c2B = F(idx++);
    const float* c3W = F(idx++); const float* c3B = F(idx++);
    const float* g1W = F(idx++); const float* g1B = F(idx++);
    const float* g2W = F(idx++); const float* g2B = F(idx++);

    // carve workspace
    char* wsp = (char*)d_ws;
    auto carve = [&](size_t bytes) {
        void* p = wsp;
        wsp += (bytes + 255) & ~(size_t)255;
        return p;
    };
    float*    hbuf  = (float*)carve((size_t)N * 128 * 4);
    float*    tmp1  = (float*)carve((size_t)N * 128 * 4);
    float*    tmp2  = (float*)carve((size_t)N * 128 * 4);
    float*    deg   = (float*)carve((size_t)N * 4);
    float*    ssrc  = (float*)carve((size_t)N * 4 * 4);
    float*    sdst  = (float*)carve((size_t)N * 4 * 4);
    float*    denom = (float*)carve((size_t)N * 4 * 4);
    float*    gsum  = (float*)carve(128 * 4);
    _Float16* wpack = (_Float16*)carve(128 * 128 * 2);

    const int TB = 256;
#define GRID(nthr) <<<((nthr) + TB - 1) / TB, TB, 0, stream>>>
    const int rowTiles = (N + 15) / 16;
    const int gblocks  = (rowTiles + 3) / 4;

    // projection
    k_proj GRID(N * 128)(x, pW, pB, pG, pBb, hbuf, N);

    for (int i = 0; i < 4; ++i) {
        if ((i & 1) == 0) {  // GCN
            k_zero GRID(N)(deg, N);
            k_deg  GRID(E)(dstp, deg, E);
            k_norm GRID(N)(deg, N);                      // deg -> norm in place
            k_pack_w GRID(8 * 2048)(lW[i], wpack, 8, 128);
            k_gemm128<8><<<gblocks, 128, 0, stream>>>(hbuf, wpack, lB[i],
                nullptr, nullptr, deg, tmp1, N, 0);      // (xW+b)*norm
            k_zero GRID(N * 128)(tmp2, N * 128);
            k_gcn_scatter GRID(E * 32)(src, dstp, tmp1, deg, tmp2, E);
            k_post GRID(N * 128)(tmp2, lG[i], lBb[i], hbuf, N);
        } else {             // GAT
            k_pack_w GRID(8 * 2048)(lW[i], wpack, 8, 128);
            k_gemm128<8><<<gblocks, 128, 0, stream>>>(hbuf, wpack, lB[i],
                nullptr, nullptr, nullptr, tmp1, N, 0);  // xW+b
            k_gat_scores GRID(N * 4)(tmp1, lAs[i], lAd[i], ssrc, sdst, N);
            k_zero GRID(N * 4)(denom, N * 4);
            k_gat_edge1 GRID(E * 4)(src, dstp, ssrc, sdst, denom, E);
            k_zero GRID(N * 128)(tmp2, N * 128);
            k_gat_edge2 GRID(E * 32)(src, dstp, tmp1, ssrc, sdst, denom, tmp2, E);
            k_post GRID(N * 128)(tmp2, lG[i], lBb[i], hbuf, N);
        }
    }

    // classifier
    float* out = (float*)d_out;
    k_pack_w GRID(8 * 2048)(c1W, wpack, 8, 128);
    k_gemm128<8><<<gblocks, 128, 0, stream>>>(hbuf, wpack, c1B, cbG, cbB,
                                              nullptr, tmp1, N, 1);   // relu(bn(.))
    k_pack_w GRID(4 * 2048)(c2W, wpack, 4, 64);
    k_gemm128<4><<<gblocks, 128, 0, stream>>>(tmp1, wpack, c2B, nullptr, nullptr,
                                              nullptr, tmp2, N, 1);   // relu(.)
    k_cls3 GRID(N)(tmp2, c3W, c3B, out, N);

    // graph score
    k_zero<<<1, 128, 0, stream>>>(gsum, 128);
    k_mean<<<(N + 63) / 64, 128, 0, stream>>>(hbuf, gsum, N);
    k_gshead<<<1, 64, 0, stream>>>(gsum, g1W, g1B, g2W, g2B, out + (size_t)N * 3, N);
#undef GRID
}
